// AlternatingHighwayLSTM_46110768889878
// MI455X (gfx1250) — matrix operand order repaired and checked by reading, not compiled, not run on hardware
//
#include <hip/hip_runtime.h>

// =====================================================================
// AlternatingHighwayLSTM for MI455X (gfx1250, wave32, WMMA)
//
// Phase 1 (parallel, WMMA): XW[t, 0:1536] = x_t @ [W_ih; Wg_x; Whw]^T + bias
//   -> v_wmma_f32_16x16x32_f16, one 16x16 tile per wave, f32 accumulate.
// Phase 2 (serial scan): 1 workgroup, 1024 lanes (32 wave32), W_hh row
//   per lane in 128 packed-f16 VGPRs, Wgh quarter-row in 32 more; h in
//   LDS read wave-uniform (broadcast); v_pk_fma_f16 matvec; gate staging
//   + activations between two barriers; next row prefetched.
// =====================================================================

typedef __attribute__((ext_vector_type(16))) _Float16 v16h;
typedef __attribute__((ext_vector_type(8)))  float    v8f;
typedef __attribute__((ext_vector_type(2)))  _Float16 h2;

#define HDIM 256
#define NGATE 1024
#define NTOT 1536   // 1024 gate + 256 gx + 256 hw columns

__device__ __forceinline__ float sigmoidf_(float x) {
    return 1.0f / (1.0f + __expf(-x));
}

// ---------------- prep kernels ----------------
__global__ void k_f32_to_f16(const float* __restrict__ s, _Float16* __restrict__ d, int n) {
    int i = blockIdx.x * blockDim.x + threadIdx.x;
    if (i < n) d[i] = (_Float16)s[i];
}

// Wcat rows: [0,1024) = W_ih, [1024,1280) = Wg[:, H:], [1280,1536) = Whw
__global__ void k_build_wcat(_Float16* __restrict__ dst,
                             const float* __restrict__ W_ih,
                             const float* __restrict__ Wg,
                             const float* __restrict__ Whw,
                             int K, int wg_stride, int wg_off) {
    int i = blockIdx.x * blockDim.x + threadIdx.x;
    if (i >= NTOT * K) return;
    int r = i / K, k = i - r * K;
    float v;
    if (r < NGATE)            v = W_ih[(size_t)r * K + k];
    else if (r < NGATE + HDIM) v = Wg[(size_t)(r - NGATE) * wg_stride + wg_off + k];
    else                       v = Whw[(size_t)(r - NGATE - HDIM) * K + k];
    dst[i] = (_Float16)v;
}

__global__ void k_build_bias(float* __restrict__ dst,
                             const float* __restrict__ b_ih,
                             const float* __restrict__ b_hh,
                             const float* __restrict__ bg) {
    int i = blockIdx.x * blockDim.x + threadIdx.x;
    if (i >= NTOT) return;
    float v;
    if (i < NGATE)            v = b_ih[i] + b_hh[i];
    else if (i < NGATE + HDIM) v = bg[i - NGATE];
    else                       v = 0.0f;
    dst[i] = v;
}

// Wgh = Wg[:, :H] as f16, [256 x 256]
__global__ void k_build_wgh(_Float16* __restrict__ dst,
                            const float* __restrict__ Wg, int wg_stride) {
    int i = blockIdx.x * blockDim.x + threadIdx.x;
    if (i >= HDIM * HDIM) return;
    int r = i >> 8, c = i & 255;
    dst[i] = (_Float16)Wg[(size_t)r * wg_stride + c];
}

// ---------------- WMMA GEMM: C[t,n] = A[t,:] . B[n,:] + bias[n] ----------------
// A: [T x K] f16 row-major (row index reversed if reverse!=0)
// B: [1536 x K] f16 row-major (Wcat)
// C: [T x 1536] f32
__global__ __launch_bounds__(256)
void k_gemm_wmma(const _Float16* __restrict__ A, const _Float16* __restrict__ B,
                 const float* __restrict__ bias, float* __restrict__ C,
                 int T, int K, int reverse) {
    const int tiles_n = NTOT >> 4;                 // 96
    int wave = threadIdx.x >> 5;                   // 8 waves / block
    int lane = threadIdx.x & 31;
    long tile = (long)blockIdx.x * 8 + wave;
    int tm = (int)(tile / tiles_n);
    int tn = (int)(tile % tiles_n);
    if (tm * 16 >= T) return;

    int half = lane >> 4;                          // lane half (0/1)
    int l16  = lane & 15;
    int arow = tm * 16 + l16;
    if (reverse) arow = T - 1 - arow;
    const _Float16* ap = A + (size_t)arow * K;
    const _Float16* bp = B + (size_t)(tn * 16 + l16) * K;

    v8f acc = {};
    for (int k0 = 0; k0 < K; k0 += 32) {
        union { v16h v; float4 f[2]; } ua, ub;
        // A (16x32, f16): lane-half 0 -> K = k0+[0..7] & k0+16+[0..7]
        //                 lane-half 1 -> K = k0+8+[0..7] & k0+24+[0..7]
        ua.f[0] = *(const float4*)(ap + k0 + half * 8);
        ua.f[1] = *(const float4*)(ap + k0 + 16 + half * 8);
        // B (32x16, f16): same K layout as A (half*8 and 16+half*8)
        ub.f[0] = *(const float4*)(bp + k0 + half * 8);
        ub.f[1] = *(const float4*)(bp + k0 + 16 + half * 8);
        acc = __builtin_amdgcn_wmma_f32_16x16x32_f16(false, ua.v, false, ub.v,
                                                     (short)0, acc, false, false);
    }
    // D layout: VGPR r -> row (r + 8*half), col = l16
    float bcol = bias[tn * 16 + l16];
    float* cp = C + (size_t)(tm * 16) * NTOT + tn * 16 + l16;
#pragma unroll
    for (int r = 0; r < 8; ++r) {
        int m = r + half * 8;
        cp[(size_t)m * NTOT] = acc[r] + bcol;
    }
}

// ---------------- serial highway-LSTM scan ----------------
// XW: [T x 1536] f32 preactivations (gates | gx | hw)
// Whh: [1024 x 256] f16, Wgh: [256 x 256] f16
// out32: [T x 256] f32 (final h), out16: [T x 256] f16 (next layer input)
__global__ __launch_bounds__(1024)
void k_scan(const float* __restrict__ XW, const _Float16* __restrict__ Whh,
            const _Float16* __restrict__ Wgh,
            float* __restrict__ out32, _Float16* __restrict__ out16, int T) {
    __shared__ __align__(16) _Float16 h16[HDIM];
    __shared__ float stage[1024];
    __shared__ float wgst[1024];

    const int tid = threadIdx.x;     // 0..1023: owns gate row tid of W_hh
    const int q   = tid >> 8;        // quarter of the w-gate dot product
    const int wr  = tid & 255;       // w-gate row

    // W_hh row -> 128 packed-f16 VGPRs
    h2 w[128];
    {
        const float4* p = (const float4*)(Whh + (size_t)tid * HDIM);
#pragma unroll
        for (int i = 0; i < 32; ++i) {
            union { float4 f; h2 h[4]; } u; u.f = p[i];
            w[4*i+0] = u.h[0]; w[4*i+1] = u.h[1];
            w[4*i+2] = u.h[2]; w[4*i+3] = u.h[3];
        }
    }
    // Wgh quarter row (columns [q*64, q*64+64)) -> 32 packed-f16 VGPRs
    h2 wg[32];
    {
        const float4* p = (const float4*)(Wgh + (size_t)wr * HDIM + q * 64);
#pragma unroll
        for (int i = 0; i < 8; ++i) {
            union { float4 f; h2 h[4]; } u; u.f = p[i];
            wg[4*i+0] = u.h[0]; wg[4*i+1] = u.h[1];
            wg[4*i+2] = u.h[2]; wg[4*i+3] = u.h[3];
        }
    }

    if (tid < 32) {
        float4 z; z.x = z.y = z.z = z.w = 0.0f;
        ((float4*)h16)[tid] = z;
    }
    float c = 0.0f;
    __syncthreads();

    // prefetch step 0
    float pf_g  = XW[tid];
    float pf_gx = 0.0f, pf_hw = 0.0f;
    if (tid < HDIM) { pf_gx = XW[NGATE + tid]; pf_hw = XW[NGATE + HDIM + tid]; }

    const float4* hv4 = (const float4*)h16;
    const h2 hz = { (_Float16)0, (_Float16)0 };

    for (int t = 0; t < T; ++t) {
        float xw_g = pf_g, gx = pf_gx, hw = pf_hw;

        // gates[tid] partial: W_hh[tid,:] . h  (pk_fma_f16, broadcast LDS reads)
        h2 a0 = hz, a1 = hz, ag0 = hz, ag1 = hz;
#pragma unroll
        for (int i = 0; i < 32; ++i) {
            union { float4 f; h2 h[4]; } u; u.f = hv4[i];
            a0 = w[4*i+0] * u.h[0] + a0;
            a1 = w[4*i+1] * u.h[1] + a1;
            a0 = w[4*i+2] * u.h[2] + a0;
            a1 = w[4*i+3] * u.h[3] + a1;
            int j = i - q * 8;                 // wave-uniform branch
            if (j >= 0 && j < 8) {
                ag0 = wg[4*j+0] * u.h[0] + ag0;
                ag1 = wg[4*j+1] * u.h[1] + ag1;
                ag0 = wg[4*j+2] * u.h[2] + ag0;
                ag1 = wg[4*j+3] * u.h[3] + ag1;
            }
        }
        float acc  = (float)a0.x + (float)a0.y + (float)a1.x + (float)a1.y;
        float accg = (float)ag0.x + (float)ag0.y + (float)ag1.x + (float)ag1.y;
        stage[tid] = acc + xw_g;
        wgst[tid]  = accg;

        // prefetch next step while barriers/activations run
        {
            size_t base = (size_t)((t + 1 < T) ? t + 1 : t) * NTOT;
            pf_g = XW[base + tid];
            if (tid < HDIM) {
                pf_gx = XW[base + NGATE + tid];
                pf_hw = XW[base + NGATE + HDIM + tid];
            }
        }
        __syncthreads();

        if (tid < HDIM) {
            float gi = stage[tid];
            float gf = stage[HDIM + tid];
            float gg = stage[2 * HDIM + tid];
            float go = stage[3 * HDIM + tid];
            float wsum = wgst[tid] + wgst[256 + tid] + wgst[512 + tid] + wgst[768 + tid] + gx;
            float cn = sigmoidf_(gf) * c + sigmoidf_(gi) * tanhf(gg);
            float ho = sigmoidf_(go) * tanhf(cn);
            float wgt = sigmoidf_(wsum);
            float hn = wgt * ho + (1.0f - wgt) * hw;
            c = cn;
            h16[tid] = (_Float16)hn;
            out32[(size_t)t * HDIM + tid] = hn;
            out16[(size_t)t * HDIM + tid] = (_Float16)hn;
        }
        __syncthreads();
    }
}

// ---------------- orchestration ----------------
extern "C" void kernel_launch(void* const* d_in, const int* in_sizes, int n_in,
                              void* d_out, int out_size, void* d_ws, size_t ws_size,
                              hipStream_t stream) {
    const float* seq   = (const float*)d_in[0];
    const float* W_ih0 = (const float*)d_in[1];
    const float* W_hh0 = (const float*)d_in[2];
    const float* b_ih0 = (const float*)d_in[3];
    const float* b_hh0 = (const float*)d_in[4];
    const float* Wg0   = (const float*)d_in[5];
    const float* bg0   = (const float*)d_in[6];
    const float* Whw0  = (const float*)d_in[7];
    const float* W_ih1 = (const float*)d_in[8];
    const float* W_hh1 = (const float*)d_in[9];
    const float* b_ih1 = (const float*)d_in[10];
    const float* b_hh1 = (const float*)d_in[11];
    const float* Wg1   = (const float*)d_in[12];
    const float* bg1   = (const float*)d_in[13];
    const float* Whw1  = (const float*)d_in[14];

    const int T = in_sizes[0] / 512;   // 65536
    const int D = 512;

    char* ws = (char*)d_ws;
    size_t off = 0;
    auto alloc = [&](size_t bytes) { size_t o = off; off += (bytes + 255) & ~(size_t)255; return o; };
    size_t oX016  = alloc((size_t)T * D * 2);        // seq in f16
    size_t oXW    = alloc((size_t)T * NTOT * 4);     // preactivations f32
    size_t oH016  = alloc((size_t)T * HDIM * 2);     // layer0 h in f16
    size_t oH116  = alloc((size_t)T * HDIM * 2);     // layer1 h f16 (unused sink)
    size_t oWcat0 = alloc((size_t)NTOT * D * 2);
    size_t oWcat1 = alloc((size_t)NTOT * HDIM * 2);
    size_t oWhh0  = alloc((size_t)NGATE * HDIM * 2);
    size_t oWhh1  = alloc((size_t)NGATE * HDIM * 2);
    size_t oWgh0  = alloc((size_t)HDIM * HDIM * 2);
    size_t oWgh1  = alloc((size_t)HDIM * HDIM * 2);
    size_t oBias0 = alloc((size_t)NTOT * 4);
    size_t oBias1 = alloc((size_t)NTOT * 4);
    (void)ws_size; (void)n_in; (void)out_size;

    _Float16* X016  = (_Float16*)(ws + oX016);
    float*    XW    = (float*)(ws + oXW);
    _Float16* H016  = (_Float16*)(ws + oH016);
    _Float16* H116  = (_Float16*)(ws + oH116);
    _Float16* Wcat0 = (_Float16*)(ws + oWcat0);
    _Float16* Wcat1 = (_Float16*)(ws + oWcat1);
    _Float16* Whh0f = (_Float16*)(ws + oWhh0);
    _Float16* Whh1f = (_Float16*)(ws + oWhh1);
    _Float16* Wgh0f = (_Float16*)(ws + oWgh0);
    _Float16* Wgh1f = (_Float16*)(ws + oWgh1);
    float*    bias0 = (float*)(ws + oBias0);
    float*    bias1 = (float*)(ws + oBias1);

    auto cdiv = [](long a, long b) { return (int)((a + b - 1) / b); };

    // ---- layer 0 prep ----
    k_f32_to_f16<<<cdiv((long)T * D, 256), 256, 0, stream>>>(seq, X016, T * D);
    k_build_wcat<<<cdiv((long)NTOT * D, 256), 256, 0, stream>>>(Wcat0, W_ih0, Wg0, Whw0, D, D + HDIM, HDIM);
    k_build_bias<<<cdiv(NTOT, 256), 256, 0, stream>>>(bias0, b_ih0, b_hh0, bg0);
    k_build_wgh<<<cdiv(HDIM * HDIM, 256), 256, 0, stream>>>(Wgh0f, Wg0, D + HDIM);
    k_f32_to_f16<<<cdiv(NGATE * HDIM, 256), 256, 0, stream>>>(W_hh0, Whh0f, NGATE * HDIM);

    // ---- layer 0 GEMM (forward) + scan ----
    int gemm_blocks = (int)(((long)(T / 16) * (NTOT / 16)) / 8);
    k_gemm_wmma<<<gemm_blocks, 256, 0, stream>>>(X016, Wcat0, bias0, XW, T, D, 0);
    k_scan<<<1, 1024, 0, stream>>>(XW, Whh0f, Wgh0f, (float*)d_out /*scratch dump*/, H016, T);

    // ---- layer 1 prep ----
    k_build_wcat<<<cdiv((long)NTOT * HDIM, 256), 256, 0, stream>>>(Wcat1, W_ih1, Wg1, Whw1, HDIM, 2 * HDIM, HDIM);
    k_build_bias<<<cdiv(NTOT, 256), 256, 0, stream>>>(bias1, b_ih1, b_hh1, bg1);
    k_build_wgh<<<cdiv(HDIM * HDIM, 256), 256, 0, stream>>>(Wgh1f, Wg1, 2 * HDIM);
    k_f32_to_f16<<<cdiv(NGATE * HDIM, 256), 256, 0, stream>>>(W_hh1, Whh1f, NGATE * HDIM);

    // ---- layer 1 GEMM (reversed input) + scan -> d_out (step order) ----
    k_gemm_wmma<<<gemm_blocks, 256, 0, stream>>>(H016, Wcat1, bias1, XW, T, HDIM, 1);
    k_scan<<<1, 1024, 0, stream>>>(XW, Whh1f, Wgh1f, (float*)d_out, H116, T);
}
